// FeatExtractor_20710332301469
// MI455X (gfx1250) — matrix operand order repaired and checked by reading, not compiled, not run on hardware
//
#include <hip/hip_runtime.h>
#include <math.h>

// ---------------------------------------------------------------------------
// MI455X (gfx1250) feature extractor:
//   frames (16 x 256) per tile  --WMMA f16-split DFT-->  power spectrum
//   --> triangular bands --> log10 --> 22x22 DCT
// DFT done as dense matmul on V_WMMA_F32_16X16X32_F16 with a 3-product
// hi/lo f16 split (Ah*Bh + Ah*Bl + Al*Bh, f32 accumulate) => ~f32 accuracy.
// ---------------------------------------------------------------------------

typedef __attribute__((ext_vector_type(16))) _Float16 v16h;
typedef __attribute__((ext_vector_type(8)))  float    v8f;
typedef __attribute__((ext_vector_type(4)))  float    f4;   // trivial 16B vector

#define NHOP        128
#define NFR         256
#define NBANDS      22
#define NFRAMES_TOT 7499
#define NSAMP       960000
#define TILE_T      16
#define AST         264          // xw LDS row stride in halves (mult of 8 -> 16B aligned b128)
#define PWS         146          // pw LDS row stride (floats), avoids systematic conflicts
#define PI_F        3.14159265358979323846f

// 64 (jt,s) slots * 4 fragments * 512 halves * 2B = 256 KiB twiddle table
#define TW_HALVES   ((size_t)8 * 8 * 4 * 512)
#define TW_BYTES    (TW_HALVES * 2)

__constant__ int c_bands[NBANDS] =
    {0,4,8,12,16,20,24,28,32,40,44,48,52,56,64,72,80,88,96,104,112,128};

union Frag { v16h v; f4 q[2]; _Float16 h[16]; };

// ---------------------------------------------------------------------------
// Init kernel: twiddle fragments in WMMA B layout (contiguous k order, same as A).
// B (32x16 f16): lanes 0-15 hold K=0..15 (half h <-> K=h) for column N=lane,
// lanes 16-31 hold K=16..31 for N=lane-16.
// Fragment order per (jt,s): cos_hi, cos_lo, sin_hi, sin_lo.
// ---------------------------------------------------------------------------
__global__ void gen_twiddles(_Float16* __restrict__ tw) {
    int i = blockIdx.x * blockDim.x + threadIdx.x;
    if (i >= 8 * 8 * 32 * 16) return;               // 32768 positions
    int h    = i & 15;
    int lane = (i >> 4) & 31;
    int s    = (i >> 9) & 7;
    int jt   = (i >> 12) & 7;
    int f = jt * 16 + (lane & 15);
    int k = 32 * s + ((lane & 16) ? 16 : 0) + h;
    int p = (f * k) & 255;                          // phase mod 2*pi (256 steps)
    float ang = (float)p * (PI_F / 128.0f);
    float sn, cs;
    sincosf(ang, &sn, &cs);                         // precise; runs once per launch
    _Float16 ch = (_Float16)cs, cl = (_Float16)(cs - (float)ch);
    _Float16 sh = (_Float16)sn, sl = (_Float16)(sn - (float)sh);
    size_t base = ((size_t)(jt * 8 + s) * 4) * 512 + (size_t)lane * 16 + h;
    tw[base + 0 * 512] = ch;
    tw[base + 1 * 512] = cl;
    tw[base + 2 * 512] = sh;
    tw[base + 3 * 512] = sl;
}

// ---------------------------------------------------------------------------
// Main kernel: one WG (8 wave32) per (batch, 16-frame tile).
// ---------------------------------------------------------------------------
template <bool USE_TABLE>
__global__ __launch_bounds__(256)
void feat_kernel(const float* __restrict__ x, float* __restrict__ out,
                 const _Float16* __restrict__ tw) {
    __shared__ alignas(16) _Float16 xw_h[TILE_T * AST];   // windowed frames, hi f16
    __shared__ alignas(16) _Float16 xw_l[TILE_T * AST];   // residual, lo f16
    __shared__ float pw[TILE_T * PWS];                    // power spectrum (128 bins)
    __shared__ float bp[TILE_T * NBANDS];                 // log10 band power
    __shared__ float dctt[NBANDS * NBANDS];               // DCT-II coeffs

    const int tid  = threadIdx.x;
    const int b    = blockIdx.y;
    const int t0   = blockIdx.x * TILE_T;
    const size_t xbase = (size_t)b * NSAMP;

    // ---- Phase A: load + sqrt-Hann window + hi/lo f16 split into LDS -------
    // win[k] = sqrt(0.5*(1-cos(2*pi*k/256))) = sin(pi*k/256)
    for (int e = tid; e < TILE_T * NFR; e += 256) {
        int m = e >> 8, k = e & 255;
        int sidx = (t0 + m) * NHOP + k;
        float v = (sidx < NSAMP) ? x[xbase + sidx] : 0.0f;
        v *= __sinf((float)k * (PI_F / 256.0f));
        _Float16 hh = (_Float16)v;
        xw_h[m * AST + k] = hh;
        xw_l[m * AST + k] = (_Float16)(v - (float)hh);
    }
    // DCT table: dct[j][k] = sqrt(2/22)*cos(pi/22*(k+0.5)*j), row 0 /= sqrt(2)
    const float dscale = 0.3015113445777636f;             // sqrt(2/22)
    for (int e = tid; e < NBANDS * NBANDS; e += 256) {
        int j = e / NBANDS, k = e - j * NBANDS;
        float c = __cosf((PI_F / NBANDS) * ((float)k + 0.5f) * (float)j) * dscale;
        if (j == 0) c *= 0.70710678118654752f;
        dctt[e] = c;
    }
    __syncthreads();

    // ---- Phase B: DFT via WMMA, pw = R^2 + I^2 -----------------------------
    {
        const int lane = tid & 31;
        const int jt   = tid >> 5;                // frequency tile 0..7 (f < 128)
        const int row  = lane & 15;
        const int hi16 = (lane >> 4) & 1;
        const int aklo = hi16 ? 16 : 0;           // A layout: lanes>=16 hold K+16
        const int f    = jt * 16 + row;
        (void)f;

        v8f accr = {};
        v8f acci = {};

#pragma unroll
        for (int s = 0; s < 8; ++s) {
            Frag ah, al, bch, bcl, bsh, bsl;
            // A fragments: K block {b..b+15}, 2x b128 each plane
            const int ka = s * 32 + aklo;
            const _Float16* ph = &xw_h[row * AST + ka];
            const _Float16* pl = &xw_l[row * AST + ka];
            ah.q[0] = *(const f4*)(ph);
            ah.q[1] = *(const f4*)(ph + 8);
            al.q[0] = *(const f4*)(pl);
            al.q[1] = *(const f4*)(pl + 8);

            if (USE_TABLE) {
                // L2-resident fragment table: 2x global b128 per fragment
                const _Float16* tb =
                    tw + ((size_t)(jt * 8 + s) * 4) * 512 + (size_t)lane * 16;
                const f4* tq;
                tq = (const f4*)(tb);        bch.q[0] = tq[0]; bch.q[1] = tq[1];
                tq = (const f4*)(tb + 512);  bcl.q[0] = tq[0]; bcl.q[1] = tq[1];
                tq = (const f4*)(tb + 1024); bsh.q[0] = tq[0]; bsh.q[1] = tq[1];
                tq = (const f4*)(tb + 1536); bsl.q[0] = tq[0]; bsl.q[1] = tq[1];
            } else {
                // Fallback: build twiddles inline (integer phase reduction)
                const int kb = s * 32 + (hi16 ? 16 : 0);
#pragma unroll
                for (int hh2 = 0; hh2 < 16; ++hh2) {
                    int p = (f * (kb + hh2)) & 255;
                    float sn, cs;
                    __sincosf((float)p * (PI_F / 128.0f), &sn, &cs);
                    _Float16 c0 = (_Float16)cs;
                    bch.h[hh2] = c0;
                    bcl.h[hh2] = (_Float16)(cs - (float)c0);
                    _Float16 s0 = (_Float16)sn;
                    bsh.h[hh2] = s0;
                    bsl.h[hh2] = (_Float16)(sn - (float)s0);
                }
            }

            // 3-product split-f16 accumulate (drop Al*Bl, ~2^-22 rel. error)
            accr = __builtin_amdgcn_wmma_f32_16x16x32_f16(false, ah.v, false, bch.v, (short)0, accr, false, false);
            accr = __builtin_amdgcn_wmma_f32_16x16x32_f16(false, ah.v, false, bcl.v, (short)0, accr, false, false);
            accr = __builtin_amdgcn_wmma_f32_16x16x32_f16(false, al.v, false, bch.v, (short)0, accr, false, false);
            acci = __builtin_amdgcn_wmma_f32_16x16x32_f16(false, ah.v, false, bsh.v, (short)0, acci, false, false);
            acci = __builtin_amdgcn_wmma_f32_16x16x32_f16(false, ah.v, false, bsl.v, (short)0, acci, false, false);
            acci = __builtin_amdgcn_wmma_f32_16x16x32_f16(false, al.v, false, bsh.v, (short)0, acci, false, false);
        }

        // pw = R^2 + I^2 ; C/D layout: VGPR r <-> row r + 8*hi16, col = lane&15
        v8f pv = accr * accr + acci * acci;
        const int col = jt * 16 + row;
        const int rb  = hi16 * 8;
#pragma unroll
        for (int r = 0; r < 8; ++r) pw[(rb + r) * PWS + col] = pv[r];
    }
    __syncthreads();

    // ---- Phase C: triangular bands + log10 ---------------------------------
    for (int i = tid; i < TILE_T * NBANDS; i += 256) {
        int t = i / NBANDS, kb = i - t * NBANDS;
        const float* pr = &pw[t * PWS];
        float acc = 0.0f;
        if (kb == 0) {
            for (int ff = 0; ff < 4; ++ff) acc += (2.0f - 0.5f * ff) * pr[ff];
        } else if (kb == NBANDS - 1) {
            for (int ff = 112; ff < 128; ++ff) acc += 0.125f * (float)(ff - 112) * pr[ff];
        } else {
            int s0 = c_bands[kb - 1], m0 = c_bands[kb], e0 = c_bands[kb + 1];
            float rs = 1.0f / (float)(m0 - s0);
            float rf = 1.0f / (float)(e0 - m0);
            for (int ff = s0; ff < m0; ++ff) acc += (float)(ff - s0) * rs * pr[ff];
            for (int ff = m0; ff < e0; ++ff) acc += (1.0f - (float)(ff - m0) * rf) * pr[ff];
        }
        bp[i] = __log10f(acc);
    }
    __syncthreads();

    // ---- Phase D: 22x22 DCT + coalesced store ------------------------------
    for (int i = tid; i < TILE_T * NBANDS; i += 256) {
        int t = i / NBANDS, j = i - t * NBANDS;
        float acc = 0.0f;
#pragma unroll
        for (int k = 0; k < NBANDS; ++k) acc += bp[t * NBANDS + k] * dctt[j * NBANDS + k];
        int frame = t0 + t;
        if (frame < NFRAMES_TOT)
            out[((size_t)b * NFRAMES_TOT + frame) * NBANDS + j] = acc;
    }
}

// ---------------------------------------------------------------------------
extern "C" void kernel_launch(void* const* d_in, const int* in_sizes, int n_in,
                              void* d_out, int out_size, void* d_ws, size_t ws_size,
                              hipStream_t stream) {
    (void)n_in; (void)out_size;
    const float* x = (const float*)d_in[0];
    float* out = (float*)d_out;
    int nbatch = in_sizes[0] / NSAMP;
    if (nbatch <= 0) nbatch = 1;

    dim3 grid((NFRAMES_TOT + TILE_T - 1) / TILE_T, nbatch);

    if (d_ws != nullptr && ws_size >= TW_BYTES) {
        _Float16* tw = (_Float16*)d_ws;
        gen_twiddles<<<dim3((8 * 8 * 32 * 16 + 255) / 256), 256, 0, stream>>>(tw);
        feat_kernel<true><<<grid, 256, 0, stream>>>(x, out, tw);
    } else {
        feat_kernel<false><<<grid, 256, 0, stream>>>(x, out, nullptr);
    }
}
